// OuterEncoder_22428319220546
// MI455X (gfx1250) — matrix operand order repaired and checked by reading, not compiled, not run on hardware
//
#include <hip/hip_runtime.h>

// ---------------------------------------------------------------------------
// OuterEncoder for MI455X (gfx1250, wave32, WMMA).
// B=32, T=200, K=50, D=100, V=500000 (fixed by the reference).
//
// Roofline: gathers = ~384MB random HBM reads (~16us @ 23.3TB/s) dominate;
// GEMM work = ~19 GFLOP -> trivial on v_wmma_f32_16x16x32_f16.
// Design: persistent workgroups (800), each stages padded f16 weights into
// LDS ONCE (async direct-to-LDS, GLOBAL_LOAD_ASYNC_TO_LDS_B128), then loops
// over 8 (b,t) tiles: gather 150 embedding rows -> LDS, run both GEMMs on the
// matrix pipe entirely out of LDS, rowwise dot + softmax + weighted sum.
// ---------------------------------------------------------------------------

typedef __attribute__((ext_vector_type(16))) _Float16 v16h;
typedef __attribute__((ext_vector_type(8)))  _Float16 v8h;
typedef __attribute__((ext_vector_type(4)))  _Float16 v4h;
typedef __attribute__((ext_vector_type(8)))  float    v8f;
typedef __attribute__((ext_vector_type(4)))  int      v4i;

#if __has_builtin(__builtin_amdgcn_tanhf)
#define FAST_TANH(x) __builtin_amdgcn_tanhf(x)
#else
#define FAST_TANH(x) tanhf(x)
#endif

#define BDIM 32
#define TDIM 200
#define KTR  50
#define DDIM 100
#define NBLK (BDIM * TDIM)   // 6400 (b,t) tiles
#define GRID 800             // persistent workgroups; 8 tiles each

// WMMA-padded dims
#define MP   64     // 50 triples -> 4 m-tiles
#define NP   112    // 100 outputs -> 7 n-tiles
#define F1P  224    // 2D=200 -> 7 k-tiles of 32
#define F2P  128    // D=100  -> 4 k-tiles of 32
#define STT  113    // odd f32 row stride -> bank-conflict-free column dots

// d_ws blob: padded f16 weights + padded f32 biases, staged to LDS once.
#define SZ_WHT    (NP*F1P*2)               // 50176  f16 W_ht  [112][224] (row-major d,f)
#define SZ_WREL   (NP*F2P*2)               // 28672  f16 W_rel [112][128]
#define SZ_BIAS   (2*NP*4)                 // 896    f32 b_ht[112] + b_rel[112]
#define SZ_BLOB   (SZ_WHT + SZ_WREL + SZ_BIAS)   // 79744

// LDS layout (bytes); all offsets 16B aligned
#define OFF_WHT   0
#define OFF_WREL  (OFF_WHT + SZ_WHT)       // 50176
#define OFF_BHT   (OFF_WREL + SZ_WREL)     // 78848
#define OFF_BREL  (OFF_BHT + NP*4)         // 79296
#define OFF_HT16  (OFF_BHT + SZ_BIAS)      // 79744  f16 head_tail [64][224]
#define SZ_HT16   (MP*F1P*2)               // 28672
#define OFF_REL16 (OFF_HT16 + SZ_HT16)     // 108416 f16 relation  [64][128]
#define SZ_REL16  (MP*F2P*2)               // 16384
#define OFF_HT32  (OFF_REL16 + SZ_REL16)   // 124800 f32 head_tail [50][200]
#define SZ_HT32   (KTR*2*DDIM*4)           // 40000
#define OFF_HTT   (OFF_HT32 + SZ_HT32)     // 164800 f32 tanh(ht)  [64][113]
#define SZ_HTT    (MP*STT*4)               // 28928
#define OFF_RELT  (OFF_HTT + SZ_HTT)       // 193728 f32 rel_t     [64][113]
#define SZ_RELT   (MP*STT*4)
#define OFF_E     (OFF_RELT + SZ_RELT)     // 222656
#define OFF_ALPHA (OFF_E + 256)            // 222912
#define SMEM_TOTAL (OFF_ALPHA + 256)       // 223168 B  (< 320KB/WGP)

// ---- WMMA fragment loaders (match CDNA5 ISA §7.12.2 VGPR layouts) ----------
// A (16x32 f16, MxK): lane L holds row M=L&15; VGPR0-3 = K kb..kb+7,
// VGPR4-7 = K kb+16..kb+23, kb = (L<16 ? 0 : 8).  -> two ds_load_b128.
__device__ inline v16h load_fragA(const _Float16* p, int stride, int row0, int k0, int lane) {
  const _Float16* q = p + (row0 + (lane & 15)) * stride + k0 + ((lane >> 4) << 3);
  v8h lo = *(const v8h*)q;
  v8h hi = *(const v8h*)(q + 16);
  v16h r;
#pragma unroll
  for (int i = 0; i < 8; ++i) { r[i] = lo[i]; r[i + 8] = hi[i]; }
  return r;
}
// B (32x16 f16, KxN): lane L holds col N=L&15; same K order as A:
// VGPR0-3 = K kb..kb+7, VGPR4-7 = K kb+16..kb+23, kb = (L<16 ? 0 : 8).
// We keep W row-major as W[n][k].  -> two ds_load_b128.
__device__ inline v16h load_fragB(const _Float16* p, int stride, int n0, int k0, int lane) {
  const _Float16* q = p + (n0 + (lane & 15)) * stride + k0 + ((lane >> 4) << 3);
  v8h lo = *(const v8h*)q;
  v8h hi = *(const v8h*)(q + 16);
  v16h r;
#pragma unroll
  for (int i = 0; i < 8; ++i) { r[i] = lo[i]; r[i + 8] = hi[i]; }
  return r;
}

// ---- prep: convert + zero-pad weights/biases into d_ws once ----------------
__global__ void prep_weights(const float* __restrict__ Wht,
                             const float* __restrict__ Wrel,
                             const float* __restrict__ bht,
                             const float* __restrict__ brel,
                             char* __restrict__ blob) {
  int i = blockIdx.x * 256 + threadIdx.x;
  _Float16* wh = (_Float16*)blob;
  if (i < NP * F1P) {
    int d = i / F1P, f = i % F1P;
    wh[i] = (_Float16)((d < DDIM && f < 2 * DDIM) ? Wht[d * (2 * DDIM) + f] : 0.f);
  } else if (i < NP * F1P + NP * F2P) {
    int j = i - NP * F1P;
    int d = j / F2P, f = j % F2P;
    wh[i] = (_Float16)((d < DDIM && f < DDIM) ? Wrel[d * DDIM + f] : 0.f);
  } else if (i < NP * F1P + NP * F2P + 2 * NP) {
    int j = i - (NP * F1P + NP * F2P);
    float v;
    if (j < NP) v = (j < DDIM) ? bht[j] : 0.f;
    else        v = (j - NP < DDIM) ? brel[j - NP] : 0.f;
    ((float*)(blob + OFF_BHT))[j] = v;
  }
}

// ---- main: persistent workgroups, 8 (b,t) tiles each -----------------------
__global__ __launch_bounds__(256) void outer_encoder_kernel(
    const int* __restrict__ ids,          // [B,T,K,3]
    const float* __restrict__ emb,        // [V,D]
    const char* __restrict__ wblob,       // padded f16 weights + f32 biases
    float* __restrict__ out)              // [B,T,2D]
{
  extern __shared__ char smem[];
  _Float16* WHT   = (_Float16*)(smem + OFF_WHT);
  _Float16* WREL  = (_Float16*)(smem + OFF_WREL);
  float*    BHT   = (float*)(smem + OFF_BHT);
  float*    BREL  = (float*)(smem + OFF_BREL);
  _Float16* HT16  = (_Float16*)(smem + OFF_HT16);
  _Float16* REL16 = (_Float16*)(smem + OFF_REL16);
  float*    HT32  = (float*)(smem + OFF_HT32);
  float*    HTT   = (float*)(smem + OFF_HTT);
  float*    RELT  = (float*)(smem + OFF_RELT);
  float*    E     = (float*)(smem + OFF_E);
  float*    ALPHA = (float*)(smem + OFF_ALPHA);

  const int tid  = threadIdx.x;
  const int lane = tid & 31;
  const int wid  = tid >> 5;

  // --- ONE-TIME phase: stage weights+biases, zero the A-operand pads -------
#if __has_builtin(__builtin_amdgcn_global_load_async_to_lds_b128)
  {
    // Async direct-to-LDS copy (GLOBAL_LOAD_ASYNC_TO_LDS_B128, ASYNCcnt).
    // Builtin wants (int4 AS1*, int4 AS3*, imm offset, imm cpol); the AS3
    // cast must be an addrspacecast (strips the LDS aperture bits).
    typedef __attribute__((address_space(1))) v4i* gptr_t;
    typedef __attribute__((address_space(3))) v4i* lptr_t;
    char* wb = const_cast<char*>(wblob);
    for (int i = tid; i < SZ_BLOB / 16; i += 256) {
      __builtin_amdgcn_global_load_async_to_lds_b128(
          (gptr_t)(wb + i * 16), (lptr_t)(smem + OFF_WHT + i * 16), 0, 0);
    }
#if __has_builtin(__builtin_amdgcn_s_wait_asynccnt)
    __builtin_amdgcn_s_wait_asynccnt(0);
#else
    asm volatile("s_wait_asynccnt 0" ::: "memory");
#endif
  }
#else
  {
    const uint4* src = (const uint4*)wblob;
    uint4* dst = (uint4*)(smem + OFF_WHT);
    for (int i = tid; i < SZ_BLOB / 16; i += 256) dst[i] = src[i];
  }
#endif
  {
    uint4 z; z.x = z.y = z.z = z.w = 0u;
    uint4* dst = (uint4*)(smem + OFF_HT16);
    for (int i = tid; i < (SZ_HT16 + SZ_REL16) / 16; i += 256) dst[i] = z;
  }
  __syncthreads();

  // --- persistent loop over (b,t) tiles ------------------------------------
  for (int blk = blockIdx.x; blk < NBLK; blk += GRID) {

    // phase 1: gather 150 embedding rows (the HBM-bound part).
    // One wave per row; 25 lanes x float4 = 100 floats. f16 copy for WMMA,
    // f32 copy of head_tail for the exact alpha-weighted output sum.
    // (Pad regions of HT16/REL16 were zeroed once and are never dirtied.)
    const int* idbase = ids + (size_t)blk * KTR * 3;
    for (int r = wid; r < KTR * 3; r += 8) {
      const int k = r / 3, slot = r - k * 3;
      const int id = idbase[r];
      if (lane < 25) {
        const float4 v = *(const float4*)(emb + (size_t)id * DDIM + lane * 4);
        v4h h; h[0] = (_Float16)v.x; h[1] = (_Float16)v.y;
               h[2] = (_Float16)v.z; h[3] = (_Float16)v.w;
        if (slot == 1) {
          *(v4h*)(REL16 + k * F2P + lane * 4) = h;                // relation
        } else {
          const int col = (slot == 0) ? 0 : DDIM;                 // head | tail
          *(v4h*)(HT16 + k * F1P + col + lane * 4) = h;
          *(float4*)(HT32 + k * (2 * DDIM) + col + lane * 4) = v;
        }
      }
    }
    __syncthreads();

    // phase 2: both GEMMs on the matrix pipe, 28 tiles over 8 waves.
    for (int tile = wid; tile < 28; tile += 8) {
      const int mt = tile / 7, nt = tile % 7;
      const int n    = nt * 16 + (lane & 15);
      const int mrow = mt * 16 + ((lane >> 4) << 3);

      // GEMM1: ht_t = tanh(head_tail · W_ht^T + b_ht)   (K = 224)
      v8f acc = {};
#pragma unroll
      for (int kt = 0; kt < 7; ++kt) {
        v16h a = load_fragA(HT16, F1P, mt * 16, kt * 32, lane);
        v16h b = load_fragB(WHT,  F1P, nt * 16, kt * 32, lane);
        acc = __builtin_amdgcn_wmma_f32_16x16x32_f16(false, a, false, b,
                                                     (short)0, acc, false, false);
      }
      const float bh = BHT[n];
#pragma unroll
      for (int r = 0; r < 8; ++r)
        HTT[(mrow + r) * STT + n] = FAST_TANH(acc[r] + bh);   // v_tanh_f32

      // GEMM2: rel_t = relation · W_rel^T + b_rel       (K = 128)
      v8f acc2 = {};
#pragma unroll
      for (int kt = 0; kt < 4; ++kt) {
        v16h a = load_fragA(REL16, F2P, mt * 16, kt * 32, lane);
        v16h b = load_fragB(WREL,  F2P, nt * 16, kt * 32, lane);
        acc2 = __builtin_amdgcn_wmma_f32_16x16x32_f16(false, a, false, b,
                                                      (short)0, acc2, false, false);
      }
      const float br = BREL[n];
#pragma unroll
      for (int r = 0; r < 8; ++r)
        RELT[(mrow + r) * STT + n] = acc2[r] + br;
    }
    __syncthreads();

    // phase 3: e[k] = <rel_t[k], ht_t[k]>  (stride 113 -> conflict-free)
    if (tid < KTR) {
      float s = 0.f;
      for (int d = 0; d < DDIM; ++d)
        s += HTT[tid * STT + d] * RELT[tid * STT + d];
      E[tid] = s;
    }
    __syncthreads();

    // phase 4: softmax over K=50 (tiny; serial on thread 0)
    if (tid == 0) {
      float mx = -1e30f;
      for (int k = 0; k < KTR; ++k) mx = fmaxf(mx, E[k]);
      float s = 0.f;
      for (int k = 0; k < KTR; ++k) {
        // exp(x) = exp2(x*log2e) -> single v_exp_f32 (args <= 0, safe)
        float e = __builtin_amdgcn_exp2f((E[k] - mx) * 1.44269504088896340736f);
        ALPHA[k] = e; s += e;
      }
      const float inv = __builtin_amdgcn_rcpf(s);   // v_rcp_f32
      for (int k = 0; k < KTR; ++k) ALPHA[k] *= inv;
    }
    __syncthreads();

    // phase 5: out[f] = sum_k alpha[k] * head_tail[k][f]  (f32 exact)
    if (tid < 2 * DDIM) {
      float acc = 0.f;
      for (int k = 0; k < KTR; ++k)
        acc = fmaf(ALPHA[k], HT32[k * (2 * DDIM) + tid], acc);
      out[(size_t)blk * (2 * DDIM) + tid] = acc;
    }
    __syncthreads();   // LDS reused by next (b,t) iteration's gather
  }
}

extern "C" void kernel_launch(void* const* d_in, const int* in_sizes, int n_in,
                              void* d_out, int out_size, void* d_ws, size_t ws_size,
                              hipStream_t stream) {
  // setup_inputs order: batch_size, ids, T, emb, W_ht, b_ht, W_rel, b_rel
  const int*   ids   = (const int*)d_in[1];
  const float* emb   = (const float*)d_in[3];
  const float* W_ht  = (const float*)d_in[4];
  const float* b_ht  = (const float*)d_in[5];
  const float* W_rel = (const float*)d_in[6];
  const float* b_rel = (const float*)d_in[7];
  char* blob = (char*)d_ws;                  // 79,744 B padded weights+biases

  const int total = NP * F1P + NP * F2P + 2 * NP;   // 39648 pad elements
  prep_weights<<<(total + 255) / 256, 256, 0, stream>>>(W_ht, W_rel, b_ht, b_rel, blob);

  outer_encoder_kernel<<<GRID, 256, SMEM_TOTAL, stream>>>(
      ids, emb, blob, (float*)d_out);
}
